// TEL_58188216926925
// MI455X (gfx1250) — matrix operand order repaired and checked by reading, not compiled, not run on hardware
//
#include <hip/hip_runtime.h>
#include <hip/hip_bf16.h>

typedef __attribute__((ext_vector_type(16))) _Float16 v16h;
typedef __attribute__((ext_vector_type(8)))  float    v8f;

#define B_TOT   32768
#define D_DIM   1024
#define T_TREES 8
#define NLEAF   64
#define NNODE   63
#define O_DIM   128
#define NCOL    512           // T_TREES * NLEAF (padded from 63 -> 64 per tree)
#define ROWS_PER_BLOCK 128    // 8 waves * 16 rows
#define DEPTH   6

// ---------------------------------------------------------------------------
// smooth_step (SMOOTH_STEP_PARAM == 1.0)
// ---------------------------------------------------------------------------
__device__ __forceinline__ float smooth_step_f(float u) {
    float cubic = (-2.0f * u * u + 1.5f) * u + 0.5f;   // -2u^3 + 1.5u + 0.5
    cubic = (u <= -0.5f) ? 0.0f : cubic;
    cubic = (u >=  0.5f) ? 1.0f : cubic;
    return cubic;
}

// ---------------------------------------------------------------------------
// A-fragment (16x32, f16) from row-major f32 global matrix, converting on load.
// ISA layout: lanes 0-15 hold M=lane, K in {0..7, 16..23}; lanes 16-31 hold
// M=lane-16, K in {8..15, 24..31}.  (K = (e<8 ? e : e+8) + 8*(lane>>4))
// ---------------------------------------------------------------------------
__device__ __forceinline__ v16h load_a_f32(const float* __restrict__ base,
                                           int row0, int ld, int kb, int lane) {
    int m = lane & 15;
    int h = lane >> 4;
    const float* p = base + (size_t)(row0 + m) * ld + kb + 8 * h;
    v16h a;
#pragma unroll
    for (int e = 0; e < 8; ++e) a[e]     = (_Float16)p[e];
#pragma unroll
    for (int e = 0; e < 8; ++e) a[8 + e] = (_Float16)p[16 + e];
    return a;
}

// Same A layout but sourced from row-major f16 LDS.
__device__ __forceinline__ v16h load_a_lds(const _Float16* __restrict__ base,
                                           int row0, int ld, int kb, int lane) {
    int m = lane & 15;
    int h = lane >> 4;
    const _Float16* p = base + (row0 + m) * ld + kb + 8 * h;
    v16h a;
#pragma unroll
    for (int e = 0; e < 8; ++e) a[e]     = p[e];
#pragma unroll
    for (int e = 0; e < 8; ++e) a[8 + e] = p[16 + e];
    return a;
}

// ---------------------------------------------------------------------------
// B-fragment (32x16, f16) from a column-major f16 matrix (Wcm[col*ld + k]).
// ISA layout: lane = column N (lane&15); lanes 0-15 hold K in {0..7, 16..23},
// lanes 16-31 hold K in {8..15, 24..31}  (K = (e<8 ? e : e+8) + 8*(lane>>4)).
// Per-lane read is two 16-byte chunks -> b128-class loads.
// ---------------------------------------------------------------------------
__device__ __forceinline__ v16h load_b_cm(const _Float16* __restrict__ base,
                                          int col0, int ld, int kb, int lane) {
    int n = lane & 15;
    int h = lane >> 4;
    const _Float16* p = base + (size_t)(col0 + n) * ld + kb + 8 * h;
    v16h b;
#pragma unroll
    for (int e = 0; e < 8; ++e) b[e]     = p[e];
#pragma unroll
    for (int e = 0; e < 8; ++e) b[8 + e] = p[16 + e];
    return b;
}

__device__ __forceinline__ v8f wmma_f16(v16h a, v16h b, v8f c) {
    return __builtin_amdgcn_wmma_f32_16x16x32_f16(
        /*neg_a=*/false, a, /*neg_b=*/false, b,
        /*c_mod=*/(short)0, c, /*reuse_a=*/false, /*reuse_b=*/false);
}

// ---------------------------------------------------------------------------
// One-time repack: node_weights (T,D,63) f32 -> Wp f16, column-major padded:
// Wp[(t*64+i)*1024 + d] = nw[t][d][i], pad column i==63 is zero.
// ---------------------------------------------------------------------------
__global__ void tel_repack_node(const float* __restrict__ nw,
                                _Float16* __restrict__ Wp) {
    int idx = blockIdx.x * 256 + threadIdx.x;     // 512*1024 elements
    if (idx >= NCOL * D_DIM) return;
    int c = idx >> 10;          // padded column 0..511
    int d = idx & (D_DIM - 1);
    int t = c >> 6;
    int i = c & 63;
    float v = 0.0f;
    if (i < NNODE)
        v = nw[(size_t)t * D_DIM * NNODE + (size_t)d * NNODE + i];
    Wp[(size_t)c * D_DIM + d] = (_Float16)v;
}

// leaf_weights (T,O,64) f32 -> Lp f16 column-major by output:
// Lp[o*512 + t*64 + l] = lw[t][o][l]
__global__ void tel_repack_leaf(const float* __restrict__ lw,
                                _Float16* __restrict__ Lp) {
    int idx = blockIdx.x * 256 + threadIdx.x;     // 128*512 elements
    if (idx >= O_DIM * NCOL) return;
    int o = idx >> 9;
    int k = idx & (NCOL - 1);
    int t = k >> 6;
    int l = k & 63;
    Lp[idx] = (_Float16)lw[(size_t)t * O_DIM * NLEAF + (size_t)o * NLEAF + l];
}

// ---------------------------------------------------------------------------
// Fused kernel: GEMM1(WMMA) + smooth_step -> s(LDS f16) -> in-place tree
// expansion (mu overwrites s) -> GEMM2(WMMA) -> out.
// Block: 256 threads = 8 waves; 128 rows of B per block; grid = 256 blocks.
// Dynamic LDS: one f16 buffer [128][512] = 128 KB  (2 blocks / WGP).
// ---------------------------------------------------------------------------
__global__ __launch_bounds__(256)
void tel_fused(const float* __restrict__ x,
               const _Float16* __restrict__ Wp,
               const _Float16* __restrict__ Lp,
               float* __restrict__ out) {
    extern __shared__ _Float16 s_lds[];               // [128][512] s, then mu

    const int tid  = threadIdx.x;
    const int wave = tid >> 5;
    const int lane = tid & 31;
    const int row0 = blockIdx.x * ROWS_PER_BLOCK + wave * 16;  // global B row

    const int nlo = lane & 15;
    const int nhi = lane >> 4;

    // ---------------- Phase 1: s = smooth_step(x @ Wnode) ----------------
    // Each wave: 16 rows x 512 cols, processed as 4 passes of 128 cols
    // (8 WMMA tiles per pass) so the x A-fragments are reloaded only 4x.
#pragma unroll 1
    for (int cg = 0; cg < 4; ++cg) {
        const int c0 = cg * 128;
        v8f acc[8] = {v8f{}, v8f{}, v8f{}, v8f{}, v8f{}, v8f{}, v8f{}, v8f{}};
#pragma unroll 1
        for (int kt = 0; kt < D_DIM / 32; ++kt) {
            const int kb = kt * 32;
            v16h a = load_a_f32(x, row0, D_DIM, kb, lane);
#pragma unroll
            for (int j = 0; j < 8; ++j) {
                v16h b = load_b_cm(Wp, c0 + j * 16, D_DIM, kb, lane);
                acc[j] = wmma_f16(a, b, acc[j]);
            }
        }
        // C/D layout: VGPR v -> row v + 8*(lane>>4); col = lane&15
#pragma unroll
        for (int j = 0; j < 8; ++j) {
#pragma unroll
            for (int v = 0; v < 8; ++v) {
                int r = wave * 16 + v + 8 * nhi;
                int c = c0 + j * 16 + nlo;
                s_lds[r * NCOL + c] = (_Float16)smooth_step_f(acc[j][v]);
            }
        }
    }
    __syncthreads();

    // ---------------- Phase 2a: tree expansion, mu overwrites s ----------------
    // 128 rows * 8 trees = 1024 (b,t) pairs; 4 pairs per thread.  Each 64-entry
    // slot is read (63 s values) and then written (64 mu values) by exactly one
    // thread, so the expansion is done in place.
#pragma unroll 1
    for (int q = 0; q < 4; ++q) {
        const int pair = tid + 256 * q;
        const int bl = pair >> 3;       // local row 0..127
        const int t  = pair & 7;
        _Float16* slot = s_lds + bl * NCOL + t * NLEAF;
        float mu[NLEAF];
        mu[0] = 1.0f;
#pragma unroll
        for (int lvl = 0; lvl < DEPTH; ++lvl) {
            const int width = 1 << lvl;
            const int start = width - 1;        // 0,1,3,7,15,31
#pragma unroll
            for (int i = width - 1; i >= 0; --i) {
                float sv = (float)slot[start + i];
                float m  = mu[i];
                mu[2 * i]     = m * sv;
                mu[2 * i + 1] = m * (1.0f - sv);
            }
        }
#pragma unroll
        for (int l = 0; l < NLEAF; ++l) slot[l] = (_Float16)mu[l];
    }
    __syncthreads();

    // ---------------- Phase 2b: out = mu @ leaf^T (512 -> 128) ----------------
    v8f acc2[8] = {v8f{}, v8f{}, v8f{}, v8f{}, v8f{}, v8f{}, v8f{}, v8f{}};
#pragma unroll 1
    for (int kt = 0; kt < NCOL / 32; ++kt) {
        const int kb = kt * 32;
        v16h a = load_a_lds(s_lds, wave * 16, NCOL, kb, lane);   // mu lives here now
#pragma unroll
        for (int j = 0; j < 8; ++j) {
            v16h b = load_b_cm(Lp, j * 16, NCOL, kb, lane);
            acc2[j] = wmma_f16(a, b, acc2[j]);
        }
    }
#pragma unroll
    for (int j = 0; j < 8; ++j) {
#pragma unroll
        for (int v = 0; v < 8; ++v) {
            int r = row0 + v + 8 * nhi;
            int c = j * 16 + nlo;
            out[(size_t)r * O_DIM + c] = acc2[j][v];
        }
    }
}

// ---------------------------------------------------------------------------
extern "C" void kernel_launch(void* const* d_in, const int* in_sizes, int n_in,
                              void* d_out, int out_size, void* d_ws, size_t ws_size,
                              hipStream_t stream) {
    (void)in_sizes; (void)n_in; (void)out_size; (void)ws_size;
    const float* x  = (const float*)d_in[0];
    const float* nw = (const float*)d_in[1];   // (8, 1024, 63)
    const float* lw = (const float*)d_in[2];   // (8, 128, 64)
    float* out = (float*)d_out;                // (32768, 128)

    _Float16* Wp = (_Float16*)d_ws;            // 512*1024 f16 = 1 MB
    _Float16* Lp = Wp + (size_t)NCOL * D_DIM;  // 128*512 f16 = 128 KB

    tel_repack_node<<<(NCOL * D_DIM + 255) / 256, 256, 0, stream>>>(nw, Wp);
    tel_repack_leaf<<<(O_DIM * NCOL + 255) / 256, 256, 0, stream>>>(lw, Lp);

    const size_t lds_bytes = (size_t)ROWS_PER_BLOCK * NCOL * sizeof(_Float16); // 128 KB
    tel_fused<<<B_TOT / ROWS_PER_BLOCK, 256, lds_bytes, stream>>>(x, Wp, Lp, out);
}
